// xLSTMModel_56135222558781
// MI455X (gfx1250) — compile-verified, matrix operand order checked by reading, not run on hardware
//
#include <hip/hip_runtime.h>
#include <hip/hip_bf16.h>

typedef __attribute__((ext_vector_type(16))) __bf16 v16bf;
typedef __attribute__((ext_vector_type(8)))  float  v8f;
typedef __attribute__((ext_vector_type(4)))  int    v4i;

// pointer-to-v4i in global (AS1) and LDS (AS3) address spaces
typedef __attribute__((address_space(1))) v4i* gv4i_p;
typedef __attribute__((address_space(3))) v4i* lv4i_p;

#define TSTEPS   1024
#define BATCH    32
#define DIN      512
#define HID      1024
#define G_BLOCKS 64        // 64 WGs x 16 hidden units = 1024
#define XKT      16        // 512/32 K-tiles of x
#define HKT      32        // 1024/32 K-tiles of h
#define NKT0     48        // (512+1024)/32
#define NKT1     64        // 2048/32
#define HB_ELEM  32768     // one packed h buffer: 32x1024 bf16

// ---------- helpers ----------
__device__ __forceinline__ unsigned short f2bf(float x) {
    unsigned u = __float_as_uint(x);
    unsigned r = u + 0x7FFFu + ((u >> 16) & 1u);   // round-to-nearest-even
    return (unsigned short)(r >> 16);
}

__device__ __forceinline__ float sigf(float x) { return 1.0f / (1.0f + __expf(-x)); }

// A/B fragment slot map for 16-bit WMMA operands (16x32 / 32x16):
//   lanes 0-15 : slots 0..7 -> k 0..7,  slots 8..15 -> k 16..23
//   lanes 16-31: slots 0..7 -> k 8..15, slots 8..15 -> k 24..31
__device__ __forceinline__ int kk_of(int laneHi, int slot) {
    return (slot & 7) + (laneHi << 3) + ((slot >> 3) << 4);
}

// scatter address of element (m, j) inside an A-packed h buffer
// layout: [ktile][mtile][lane][slot], 16 bf16 contiguous per lane
__device__ __forceinline__ int hpack_addr(int m, int j) {
    int kt     = j >> 5;
    int kk     = j & 31;
    int laneHi = (kk >> 3) & 1;
    int slot   = (kk & 7) + ((kk >> 4) << 3);
    int lane   = (m & 15) + (laneHi << 4);
    int mt     = m >> 4;
    return ((kt * 2 + mt) * 32 + lane) * 16 + slot;
}

// ---------- async global -> LDS staging (CDNA5 ASYNCcnt path) ----------
#if __has_builtin(__builtin_amdgcn_global_load_async_to_lds_b128)
#define HAVE_ASYNC_LDS 1
#endif

__device__ __forceinline__ void stage_to_lds(const unsigned short* __restrict__ g,
                                             unsigned short* __restrict__ l,
                                             int nvec /*16B vectors*/, int tid) {
#if defined(HAVE_ASYNC_LDS)
    unsigned short* gm = const_cast<unsigned short*>(g);
    for (int i = tid; i < nvec; i += 256) {
        __builtin_amdgcn_global_load_async_to_lds_b128(
            (gv4i_p)(gm + (size_t)i * 8),
            (lv4i_p)(l + (size_t)i * 8),
            0, 0);
    }
#else
    for (int i = tid; i < nvec; i += 256) {
        ((uint4*)l)[i] = ((const uint4*)g)[i];
    }
#endif
}

__device__ __forceinline__ void wait_stage() {
#if defined(HAVE_ASYNC_LDS)
    asm volatile("s_wait_asynccnt 0x0" ::: "memory");
#endif
    __syncthreads();
}

// device-wide barrier (counter + generation), agent-scope fences so h vectors
// written before arrival are visible to all WGPs after release.
__device__ __forceinline__ void grid_barrier(unsigned* cnt, unsigned* gen,
                                             unsigned* sGen, unsigned nblk) {
    __builtin_amdgcn_fence(__ATOMIC_RELEASE, "agent");
    __syncthreads();
    if (threadIdx.x == 0) {
        unsigned g = *sGen;
        unsigned old = __hip_atomic_fetch_add(cnt, 1u, __ATOMIC_ACQ_REL,
                                              __HIP_MEMORY_SCOPE_AGENT);
        if (old == nblk - 1u) {
            __hip_atomic_store(cnt, 0u, __ATOMIC_RELAXED, __HIP_MEMORY_SCOPE_AGENT);
            __hip_atomic_fetch_add(gen, 1u, __ATOMIC_RELEASE, __HIP_MEMORY_SCOPE_AGENT);
        } else {
            while (__hip_atomic_load(gen, __ATOMIC_ACQUIRE,
                                     __HIP_MEMORY_SCOPE_AGENT) == g) {
                __builtin_amdgcn_s_sleep(1);
            }
        }
        *sGen = g + 1u;
    }
    __syncthreads();
    __builtin_amdgcn_fence(__ATOMIC_ACQUIRE, "agent");
}

// ---------- one-time packing kernels ----------
// Wpack[ntile][kt][lane][slot] = bf16(W[k][n]); W row-major (K, 4096)
__global__ void pack_w_kernel(const float* __restrict__ W,
                              unsigned short* __restrict__ Wp,
                              int NKT, long total) {
    long id = (long)blockIdx.x * 256 + threadIdx.x;
    if (id >= total) return;
    int  slot   = (int)(id & 15);
    int  lane   = (int)((id >> 4) & 31);
    long rem    = id >> 9;
    int  kt     = (int)(rem % NKT);
    int  ntile  = (int)(rem / NKT);
    int  n      = ntile * 16 + (lane & 15);
    long k      = (long)kt * 32 + kk_of(lane >> 4, slot);
    Wp[id] = f2bf(W[k * 4096 + n]);
}

// Xpack[t][kt][mtile][lane][slot] = bf16(X[m][t][k]); X is (B,T,D) row-major
__global__ void pack_x_kernel(const float* __restrict__ X,
                              unsigned short* __restrict__ Xp) {
    long id = (long)blockIdx.x * 256 + threadIdx.x;   // grid sized exactly
    int slot = (int)(id & 15);
    int lane = (int)((id >> 4) & 31);
    int mt   = (int)((id >> 9) & 1);
    int kt   = (int)((id >> 10) & 15);
    int t    = (int)(id >> 14);
    int m    = mt * 16 + (lane & 15);
    int k    = kt * 32 + kk_of(lane >> 4, slot);
    Xp[id] = f2bf(X[((long)m * TSTEPS + t) * DIN + k]);
}

__global__ void init_state_kernel(unsigned short* __restrict__ h0p2,
                                  unsigned short* __restrict__ h1p2,
                                  unsigned* __restrict__ syncw) {
    int id = blockIdx.x * 256 + threadIdx.x;      // 131072 threads
    if (id < 2 * HB_ELEM)  h0p2[id] = 0;
    else                   h1p2[id - 2 * HB_ELEM] = 0;
    if (id == 0) { syncw[0] = 0u; syncw[1] = 0u; }
}

// ---------- persistent 2-layer LSTM kernel ----------
// 8 waves/WG = 4 gates x 2 K-halves; each wave carries BOTH batch-half tiles
// (2 independent WMMA accumulator chains, B loaded once per WG).
__global__ __launch_bounds__(256) void lstm_persistent(
    const unsigned short* __restrict__ xp,
    const unsigned short* __restrict__ w0p,
    const unsigned short* __restrict__ w1p,
    const float* __restrict__ b0,
    const float* __restrict__ b1,
    unsigned short* __restrict__ h0p2,   // 2 parity buffers
    unsigned short* __restrict__ h1p2,   // 2 parity buffers
    float* __restrict__ h1f,
    unsigned* __restrict__ syncw) {

    __shared__ __align__(32) unsigned short aLds[2 * HB_ELEM];   // 128 KB A stage
    __shared__ float    zbuf[2][4][2][16][16];                   // [kh][gate][mt] 16 KB
    __shared__ unsigned sGen;

    const int tid   = threadIdx.x;
    const int lane  = tid & 31;
    const int wave  = tid >> 5;
    const int gate  = wave & 3;     // 0=i 1=f 2=g 3=o
    const int kh    = wave >> 2;    // K-half
    const int wg    = blockIdx.x;
    const int ntile = gate * 64 + wg;   // global 16-wide column tile in 4096

    if (tid == 0) sGen = 0u;
    __syncthreads();

    // gate-math: each thread owns elements e and e+256 of the WG's 32x16 slice
    const int j   = tid & 15;
    const int jg  = wg * 16 + j;
    const int m0  = tid >> 4;        // 0..15
    const int m1  = m0 + 16;         // 16..31
    const int ml0 = m0 & 15, ml1 = m1 & 15;
    const int hp0 = hpack_addr(m0, jg);
    const int hp1 = hpack_addr(m1, jg);

    const float bi0 = b0[jg], bf0 = b0[HID + jg], bg0 = b0[2 * HID + jg], bo0 = b0[3 * HID + jg];
    const float bi1 = b1[jg], bf1 = b1[HID + jg], bg1 = b1[2 * HID + jg], bo1 = b1[3 * HID + jg];

    float c0a = 0.f, c0b = 0.f, c1a = 0.f, c1b = 0.f;

    const unsigned short* wb0 = w0p + (size_t)ntile * NKT0 * 512;
    const unsigned short* wb1 = w1p + (size_t)ntile * NKT1 * 512;

    for (int t = 0; t < TSTEPS; ++t) {
        const int pr = (t & 1) ^ 1;            // read parity (previous step)
        const int pw = t & 1;                  // write parity (this step)
        const unsigned short* h0rd = h0p2 + (size_t)pr * HB_ELEM;
        unsigned short*       h0wr = h0p2 + (size_t)pw * HB_ELEM;
        const unsigned short* h1rd = h1p2 + (size_t)pr * HB_ELEM;
        unsigned short*       h1wr = h1p2 + (size_t)pw * HB_ELEM;

        // ====== Layer 0: z0 = [x_t, h0] @ W0  (K = 1536, 48 K-tiles) ========
        // stage A = [x_t | h0_prev] into LDS via async copies
        const unsigned short* xa = xp + (size_t)t * (XKT * 2 * 32 * 16);
        stage_to_lds(xa,   aLds,         2048, tid);   // 16 x-ktiles  (32 KB)
        stage_to_lds(h0rd, aLds + 16384, 4096, tid);   // 32 h-ktiles  (64 KB)
        wait_stage();

        v8f a0 = {}, a1 = {};
        {
            const int kt0 = kh * (NKT0 / 2);
            #pragma unroll 4
            for (int k = 0; k < NKT0 / 2; ++k) {
                int kt = kt0 + k;
                __builtin_prefetch(wb0 + (size_t)(kt + 1) * 512 + lane * 16, 0, 1);
                v16bf A0 = *(const v16bf*)(aLds + (kt * 2 + 0) * 512 + lane * 16);
                v16bf A1 = *(const v16bf*)(aLds + (kt * 2 + 1) * 512 + lane * 16);
                v16bf B  = *(const v16bf*)(wb0  + (size_t)kt * 512 + lane * 16);
                a0 = __builtin_amdgcn_wmma_f32_16x16x32_bf16(false, A0, false, B,
                                                             (short)0, a0, false, false);
                a1 = __builtin_amdgcn_wmma_f32_16x16x32_bf16(false, A1, false, B,
                                                             (short)0, a1, false, false);
            }
        }
        {   // dump both 16x16 tiles to LDS (C layout: lane%16 = n, vgpr r -> m)
            int nl = lane & 15, mh = lane >> 4;
            #pragma unroll
            for (int r = 0; r < 8; ++r) {
                zbuf[kh][gate][0][r + 8 * mh][nl] = a0[r];
                zbuf[kh][gate][1][r + 8 * mh][nl] = a1[r];
            }
        }
        __syncthreads();
        {   // layer-0 gate math (sum the two K-half partials), write h0[pw]
            float i_ = zbuf[0][0][0][ml0][j] + zbuf[1][0][0][ml0][j] + bi0;
            float f_ = zbuf[0][1][0][ml0][j] + zbuf[1][1][0][ml0][j] + bf0;
            float g_ = zbuf[0][2][0][ml0][j] + zbuf[1][2][0][ml0][j] + bg0;
            float o_ = zbuf[0][3][0][ml0][j] + zbuf[1][3][0][ml0][j] + bo0;
            c0a = sigf(f_) * c0a + sigf(i_) * tanhf(g_);
            h0wr[hp0] = f2bf(sigf(o_) * tanhf(c0a));

            i_ = zbuf[0][0][1][ml1][j] + zbuf[1][0][1][ml1][j] + bi0;
            f_ = zbuf[0][1][1][ml1][j] + zbuf[1][1][1][ml1][j] + bf0;
            g_ = zbuf[0][2][1][ml1][j] + zbuf[1][2][1][ml1][j] + bg0;
            o_ = zbuf[0][3][1][ml1][j] + zbuf[1][3][1][ml1][j] + bo0;
            c0b = sigf(f_) * c0b + sigf(i_) * tanhf(g_);
            h0wr[hp1] = f2bf(sigf(o_) * tanhf(c0b));
        }
        grid_barrier(&syncw[0], &syncw[1], &sGen, G_BLOCKS);

        // ====== Layer 1: z1 = [h0_new, h1_prev] @ W1  (K = 2048, 64 tiles) ==
        stage_to_lds(h0wr, aLds,         4096, tid);   // h0 of this step
        stage_to_lds(h1rd, aLds + 32768, 4096, tid);   // h1 of previous step
        wait_stage();

        v8f b0acc = {}, b1acc = {};
        {
            const int kt0 = kh * (NKT1 / 2);
            #pragma unroll 4
            for (int k = 0; k < NKT1 / 2; ++k) {
                int kt = kt0 + k;
                __builtin_prefetch(wb1 + (size_t)(kt + 1) * 512 + lane * 16, 0, 1);
                v16bf A0 = *(const v16bf*)(aLds + (kt * 2 + 0) * 512 + lane * 16);
                v16bf A1 = *(const v16bf*)(aLds + (kt * 2 + 1) * 512 + lane * 16);
                v16bf B  = *(const v16bf*)(wb1  + (size_t)kt * 512 + lane * 16);
                b0acc = __builtin_amdgcn_wmma_f32_16x16x32_bf16(false, A0, false, B,
                                                                (short)0, b0acc, false, false);
                b1acc = __builtin_amdgcn_wmma_f32_16x16x32_bf16(false, A1, false, B,
                                                                (short)0, b1acc, false, false);
            }
        }
        {
            int nl = lane & 15, mh = lane >> 4;
            #pragma unroll
            for (int r = 0; r < 8; ++r) {
                zbuf[kh][gate][0][r + 8 * mh][nl] = b0acc[r];
                zbuf[kh][gate][1][r + 8 * mh][nl] = b1acc[r];
            }
        }
        __syncthreads();
        {   // layer-1 gate math, write h1[pw] packed + fp32 copy for FC head
            float i_ = zbuf[0][0][0][ml0][j] + zbuf[1][0][0][ml0][j] + bi1;
            float f_ = zbuf[0][1][0][ml0][j] + zbuf[1][1][0][ml0][j] + bf1;
            float g_ = zbuf[0][2][0][ml0][j] + zbuf[1][2][0][ml0][j] + bg1;
            float o_ = zbuf[0][3][0][ml0][j] + zbuf[1][3][0][ml0][j] + bo1;
            c1a = sigf(f_) * c1a + sigf(i_) * tanhf(g_);
            float h = sigf(o_) * tanhf(c1a);
            h1wr[hp0] = f2bf(h);
            h1f[m0 * HID + jg] = h;

            i_ = zbuf[0][0][1][ml1][j] + zbuf[1][0][1][ml1][j] + bi1;
            f_ = zbuf[0][1][1][ml1][j] + zbuf[1][1][1][ml1][j] + bf1;
            g_ = zbuf[0][2][1][ml1][j] + zbuf[1][2][1][ml1][j] + bg1;
            o_ = zbuf[0][3][1][ml1][j] + zbuf[1][3][1][ml1][j] + bo1;
            c1b = sigf(f_) * c1b + sigf(i_) * tanhf(g_);
            h = sigf(o_) * tanhf(c1b);
            h1wr[hp1] = f2bf(h);
            h1f[m1 * HID + jg] = h;
        }
        grid_barrier(&syncw[0], &syncw[1], &sGen, G_BLOCKS);
    }
}

// ---------- final FC head: out[b] = h1[b,:] . fc_w + fc_b ----------
__global__ void fc_kernel(const float* __restrict__ h1f,
                          const float* __restrict__ fcw,
                          const float* __restrict__ fcb,
                          float* __restrict__ out) {
    __shared__ float red[256];
    int tid = threadIdx.x;
    int b = tid >> 3, l = tid & 7;
    float s = 0.f;
    for (int k = l; k < HID; k += 8) s += h1f[b * HID + k] * fcw[k];
    red[tid] = s;
    __syncthreads();
    if (l == 0) {
        float t = 0.f;
        #pragma unroll
        for (int q = 0; q < 8; ++q) t += red[(b << 3) + q];
        out[b] = t + fcb[0];
    }
}

extern "C" void kernel_launch(void* const* d_in, const int* in_sizes, int n_in,
                              void* d_out, int out_size, void* d_ws, size_t ws_size,
                              hipStream_t stream) {
    const float* X   = (const float*)d_in[0];   // (32,1024,512)
    const float* W0  = (const float*)d_in[1];   // (1536,4096)
    const float* b0  = (const float*)d_in[2];   // (4096,)
    const float* W1  = (const float*)d_in[3];   // (2048,4096)
    const float* b1  = (const float*)d_in[4];   // (4096,)
    const float* fcw = (const float*)d_in[5];   // (1024,1)
    const float* fcb = (const float*)d_in[6];   // (1,)
    float* out = (float*)d_out;                 // (32,1)

    char* ws = (char*)d_ws;
    unsigned*       syncw = (unsigned*)(ws);                          // 256 B
    unsigned short* h0p2  = (unsigned short*)(ws + 256);              // 128 KB (2 parities)
    unsigned short* h1p2  = (unsigned short*)(ws + 256 + 131072);     // 128 KB
    float*          h1f   = (float*)(ws + 256 + 262144);              // 128 KB
    unsigned short* w0p   = (unsigned short*)(ws + 393472);           // 12.6 MB
    unsigned short* w1p   = w0p + 6291456;                            // 16.8 MB
    unsigned short* xp    = w1p + 8388608;                            // 33.6 MB

    pack_w_kernel<<<(6291456 + 255) / 256, 256, 0, stream>>>(W0, w0p, NKT0, 6291456L);
    pack_w_kernel<<<(8388608 + 255) / 256, 256, 0, stream>>>(W1, w1p, NKT1, 8388608L);
    pack_x_kernel<<<16777216 / 256, 256, 0, stream>>>(X, xp);
    init_state_kernel<<<131072 / 256, 256, 0, stream>>>(h0p2, h1p2, syncw);

    lstm_persistent<<<G_BLOCKS, 256, 0, stream>>>(xp, w0p, w1p, b0, b1,
                                                  h0p2, h1p2, h1f, syncw);

    fc_kernel<<<1, 256, 0, stream>>>(h1f, fcw, fcb, out);
}
